// PointTransformerCPU_28063316312780
// MI455X (gfx1250) — compile-verified, matrix operand order checked by reading, not run on hardware
//
#include <hip/hip_runtime.h>
#include <math.h>

// MI455X / gfx1250 PointTransformer forward.
// All dense linear algebra -> v_wmma_f32_16x16x32_f16 (f16 operands, f32 accum).
// Weights are converted to f16 once per launch; activations flow through f16
// buffers between GEMMs (fp32 residual stream kept). Fragment loads exploit the
// CDNA5 16-bit A/B layout: per lane two contiguous 8-half runs -> b128 loads.

typedef __attribute__((ext_vector_type(16))) _Float16 v16h;
typedef __attribute__((ext_vector_type(8)))  _Float16 v8h;
typedef __attribute__((ext_vector_type(8)))  float    v8f;

namespace {

constexpr int  BB   = 16;      // batch
constexpr int  NP   = 32768;   // points
constexpr int  CIN  = 6;       // channels
constexpr int  NG   = 256;     // NUM_GROUP
constexpr int  GS   = 32;      // GROUP_SIZE
constexpr int  TD   = 384;     // TRANS_DIM
constexpr int  NH   = 6;       // heads
constexpr int  HD   = 64;      // head dim
constexpr int  SEQ  = 257;     // cls + groups
constexpr int  SEQP = 264;     // SEQ padded to multiple of 8 (f16 score rows)
constexpr int  QKVD = 3 * TD;  // 1152
constexpr float EPSF = 1e-5f;

#define DEVINL __device__ __forceinline__

DEVINL v8f wmma_f16(v16h a, v16h b, v8f c) {
  return __builtin_amdgcn_wmma_f32_16x16x32_f16(false, a, false, b, (short)0, c,
                                                false, false);
}

// Fragment layout (ISA 7.12.2): lane half h selects K+8/K+24; within a lane the
// 16 halves are two contiguous runs: k0+h+[0,8) and k0+16+h+[0,8).

// A fragment, f16 row-major [M,lda], K-run fully in bounds (K % 32 == 0).
DEVINL v16h load_a_h(const _Float16* A, long lda, int m0, int k0, int M) {
  int lane = threadIdx.x & 31;
  int m = m0 + (lane & 15);
  int h = (lane >> 4) << 3;
  v16h r = {};
  if (m < M) {
    const _Float16* p0 = A + (long)m * lda + k0 + h;
    v8h lo = *(const v8h*)p0;
    v8h hi = *(const v8h*)(p0 + 16);
#pragma unroll
    for (int e = 0; e < 8; ++e) { r[e] = lo[e]; r[e + 8] = hi[e]; }
  }
  return r;
}

// A fragment, f16 row-major, K bounded (for 257-wide softmax rows).
DEVINL v16h load_a_h_bnd(const _Float16* A, long lda, int m0, int k0, int M, int Kb) {
  int lane = threadIdx.x & 31;
  int m = m0 + (lane & 15);
  int h = (lane >> 4) << 3;
  v16h r = {};
  if (m < M) {
    const _Float16* row = A + (long)m * lda;
#pragma unroll
    for (int e = 0; e < 8; ++e) {
      int k = k0 + h + e;
      if (k < Kb) r[e] = row[k];
      int k2 = k + 16;
      if (k2 < Kb) r[e + 8] = row[k2];
    }
  }
  return r;
}

// B fragment where B[k][n] = W[n*ldw + k], W f16 [N,K] (K % 32 == 0).
DEVINL v16h load_b_nk_h(const _Float16* Wt, long ldw, int n0, int k0, int Nb) {
  int lane = threadIdx.x & 31;
  int n = n0 + (lane & 15);
  int h = (lane >> 4) << 3;
  v16h r = {};
  if (n < Nb) {
    const _Float16* p0 = Wt + (long)n * ldw + k0 + h;
    v8h lo = *(const v8h*)p0;
    v8h hi = *(const v8h*)(p0 + 16);
#pragma unroll
    for (int e = 0; e < 8; ++e) { r[e] = lo[e]; r[e + 8] = hi[e]; }
  }
  return r;
}

// B fragment where B[k][n] = V[k*ldv + n] (K-strided, scalar f16 loads), K bounded.
DEVINL v16h load_b_kn_h(const _Float16* V, long ldv, int n0, int k0, int Kb) {
  int lane = threadIdx.x & 31;
  int n = n0 + (lane & 15);
  int h = (lane >> 4) << 3;
  v16h r = {};
#pragma unroll
  for (int e = 0; e < 8; ++e) {
    int k = k0 + h + e;
    if (k < Kb) r[e] = V[(long)k * ldv + n];
    int k2 = k + 16;
    if (k2 < Kb) r[e + 8] = V[(long)k2 * ldv + n];
  }
  return r;
}

// A fragment from f16 LDS tile (in-bounds, 16B aligned rows) -> ds_load_b128.
DEVINL v16h load_a_lds(const _Float16* A, int lda, int m0, int k0) {
  int lane = threadIdx.x & 31;
  int m = m0 + (lane & 15);
  int h = (lane >> 4) << 3;
  const _Float16* p0 = A + m * lda + k0 + h;
  v8h lo = *(const v8h*)p0;
  v8h hi = *(const v8h*)(p0 + 16);
  v16h r;
#pragma unroll
  for (int e = 0; e < 8; ++e) { r[e] = lo[e]; r[e + 8] = hi[e]; }
  return r;
}

DEVINL float gelu_exact(float v) {
  return 0.5f * v * (1.0f + erff(v * 0.70710678118f));
}

DEVINL void atomicMaxF_sh(float* addr, float val) {
  int* ai = reinterpret_cast<int*>(addr);
  int cur = *ai;
  while (__int_as_float(cur) < val) {
    int prev = atomicCAS(ai, cur, __float_as_int(val));
    if (prev == cur) break;
    cur = prev;
  }
}

// -------------------------------- utility ----------------------------------
__global__ void k_zero(float* __restrict__ p, long n) {
  long i = (long)blockIdx.x * blockDim.x + threadIdx.x;
  if (i < n) p[i] = 0.0f;
}

__global__ void k_add(float* __restrict__ x, const float* __restrict__ y, long n) {
  long i = (long)blockIdx.x * blockDim.x + threadIdx.x;
  if (i < n) x[i] += y[i];
}

__global__ void k_cvt(const float* __restrict__ s, _Float16* __restrict__ d, long n) {
  long i = (long)blockIdx.x * blockDim.x + threadIdx.x;
  if (i < n) d[i] = (_Float16)s[i];
}

// --------------------------------- FPS -------------------------------------
__global__ __launch_bounds__(1024) void k_fps(const float* __restrict__ pts,
                                              float* __restrict__ dist,
                                              int* __restrict__ fidx) {
  int b = blockIdx.x;
  const float* P = pts + (long)b * NP * CIN;
  float* D = dist + (long)b * NP;
  int* out = fidx + b * NG;
  __shared__ float rv[1024];
  __shared__ int   ri[1024];
  __shared__ float fpt[3];
  __shared__ int   sfar;
  int tid = threadIdx.x;
  if (tid == 0) sfar = 0;
  for (int i = tid; i < NP; i += 1024) D[i] = 1e10f;
  __syncthreads();
  for (int step = 0; step < NG; ++step) {
    int far = sfar;
    if (tid == 0) out[step] = far;
    if (tid < 3) fpt[tid] = P[(long)far * CIN + tid];
    __syncthreads();
    float fx = fpt[0], fy = fpt[1], fz = fpt[2];
    float bmax = -1.0f; int bidx = 0;
    for (int i = tid; i < NP; i += 1024) {
      const float* q = P + (long)i * CIN;
      float dx = q[0] - fx, dy = q[1] - fy, dz = q[2] - fz;
      float dd = fminf(D[i], dx * dx + dy * dy + dz * dz);
      D[i] = dd;
      if (dd > bmax) { bmax = dd; bidx = i; }
    }
    rv[tid] = bmax; ri[tid] = bidx;
    __syncthreads();
    for (int off = 512; off > 0; off >>= 1) {
      if (tid < off) {
        float v2 = rv[tid + off]; int i2 = ri[tid + off];
        if (v2 > rv[tid] || (v2 == rv[tid] && i2 < ri[tid])) { rv[tid] = v2; ri[tid] = i2; }
      }
      __syncthreads();
    }
    if (tid == 0) sfar = ri[0];
    __syncthreads();
  }
}

// ------------------------------ kNN + gather --------------------------------
__global__ __launch_bounds__(256) void k_knn(const float* __restrict__ pts,
                                             const int* __restrict__ fidx,
                                             float* __restrict__ centers,
                                             float* __restrict__ nb) {
  int b = blockIdx.x / NG, g = blockIdx.x % NG;
  const float* P = pts + (long)b * NP * CIN;
  int tid = threadIdx.x;
  __shared__ float cen[3];
  __shared__ float rv[256];
  __shared__ int   ri[256];
  __shared__ int   sel[GS];
  if (tid == 0) {
    int ci = fidx[b * NG + g];
    cen[0] = P[(long)ci * CIN + 0];
    cen[1] = P[(long)ci * CIN + 1];
    cen[2] = P[(long)ci * CIN + 2];
  }
  __syncthreads();
  float cx = cen[0], cy = cen[1], cz = cen[2];
  float ld[NP / 256];
  for (int t = 0; t < NP / 256; ++t) {
    long i = (long)t * 256 + tid;
    const float* q = P + i * CIN;
    float dx = q[0] - cx, dy = q[1] - cy, dz = q[2] - cz;
    ld[t] = dx * dx + dy * dy + dz * dz;
  }
  for (int kk = 0; kk < GS; ++kk) {
    float bv = 1e30f; int bt = 0;
    for (int t = 0; t < NP / 256; ++t)
      if (ld[t] < bv) { bv = ld[t]; bt = t; }
    rv[tid] = bv; ri[tid] = bt * 256 + tid;
    __syncthreads();
    for (int off = 128; off > 0; off >>= 1) {
      if (tid < off) {
        float v2 = rv[tid + off]; int i2 = ri[tid + off];
        if (v2 < rv[tid] || (v2 == rv[tid] && i2 < ri[tid])) { rv[tid] = v2; ri[tid] = i2; }
      }
      __syncthreads();
    }
    int win = ri[0];
    if (tid == 0) sel[kk] = win;
    if ((win & 255) == tid) ld[win >> 8] = 1e30f;
    __syncthreads();
  }
  if (tid < 3) centers[((long)b * NG + g) * 3 + tid] = cen[tid];
  if (tid < GS * CIN) {
    int kk = tid / CIN, ch = tid % CIN;
    long pi = sel[kk];
    float v = P[pi * CIN + ch];
    if (ch < 3) v -= cen[ch];
    nb[(((long)b * NG + g) * GS + kk) * CIN + ch] = v;
  }
}

// ----------------------- encoder stage 1 (6 -> 128) -------------------------
__global__ __launch_bounds__(128) void k_enc1(const float* __restrict__ nb,
                                              const float* __restrict__ w,   // [128,6]
                                              const float* __restrict__ bias,
                                              float* __restrict__ h1,
                                              float* __restrict__ stats) {
  int ch = threadIdx.x;
  long row0 = (long)blockIdx.x * 64;
  const float* wr = w + ch * 6;
  float s = 0.f, s2 = 0.f;
  for (int r = 0; r < 64; ++r) {
    long row = row0 + r;
    const float* f = nb + row * CIN;
    float v = bias[ch];
    for (int j = 0; j < CIN; ++j) v += wr[j] * f[j];
    h1[row * 128 + ch] = v;
    s += v; s2 += v * v;
  }
  atomicAdd(&stats[ch], s);
  atomicAdd(&stats[128 + ch], s2);
}

__global__ void k_bn_finalize(float* __restrict__ st, int ch, float cnt) {
  int i = blockIdx.x * blockDim.x + threadIdx.x;
  if (i >= ch) return;
  float m = st[i] / cnt;
  float v = st[ch + i] / cnt - m * m;
  st[i] = m;
  st[ch + i] = rsqrtf(v + EPSF);
}

// ------------- encoder per-group pipeline (WMMA from LDS f16) ---------------
__global__ __launch_bounds__(256) void k_enc_group(
    const float* __restrict__ h1, const float* __restrict__ bn128,
    const float* __restrict__ g128, const float* __restrict__ b128,
    const _Float16* __restrict__ w2, const float* __restrict__ b2,   // [256,128]
    const _Float16* __restrict__ w3, const float* __restrict__ b3,   // [512,512]
    float* __restrict__ st512,                                        // mode0 accum / mode1 mean+inv
    const float* __restrict__ g512, const float* __restrict__ b512,
    const _Float16* __restrict__ w4, const float* __restrict__ b4,   // [384,512]
    float* __restrict__ tokens, int mode)
{
  __shared__ __align__(16) _Float16 a1[GS * 128];
  __shared__ __align__(16) _Float16 h2h[GS * 256];
  __shared__ __align__(16) _Float16 c2[GS * 512];
  __shared__ float gmax4[TD];
  long grp = blockIdx.x;
  const float* h1g = h1 + grp * GS * 128;
  int tid = threadIdx.x, wave = tid >> 5, lane = tid & 31;

  for (int i = tid; i < GS * 128; i += 256) {
    int ch = i & 127;
    float v = (h1g[i] - bn128[ch]) * bn128[128 + ch] * g128[ch] + b128[ch];
    a1[i] = (_Float16)fmaxf(v, 0.0f);
  }
  __syncthreads();

  // h2 = a1 @ w2^T + b2  (M=32,N=256,K=128)
  for (int t = wave; t < 32; t += 8) {
    int mt = t >> 4, nt = t & 15;
    v8f acc = {};
    for (int k0 = 0; k0 < 128; k0 += 32) {
      v16h a = load_a_lds(a1, 128, mt * 16, k0);
      v16h bf = load_b_nk_h(w2, 128, nt * 16, k0, 256);
      acc = wmma_f16(a, bf, acc);
    }
    int n = nt * 16 + (lane & 15);
    int mb = mt * 16 + ((lane >> 4) << 3);
    float bi = b2[n];
#pragma unroll
    for (int r = 0; r < 8; ++r) h2h[(mb + r) * 256 + n] = (_Float16)(acc[r] + bi);
  }
  __syncthreads();

  // per-column max over 32 rows; concat [gmax | h2] into c2
  {
    int n = tid;
    float gm = -1e30f;
    for (int r = 0; r < GS; ++r) gm = fmaxf(gm, (float)h2h[r * 256 + n]);
    _Float16 gmh = (_Float16)gm;
    for (int r = 0; r < GS; ++r) {
      c2[r * 512 + n] = gmh;
      c2[r * 512 + 256 + n] = h2h[r * 256 + n];
    }
  }
  __syncthreads();

  // h3 = c2 @ w3^T + b3  (M=32,N=512,K=512 -> 64 tiles, 8/wave)
  v8f h3acc[8];
  {
    int ti = 0;
    for (int t = wave; t < 64; t += 8, ++ti) {
      int mt = t >> 5, nt = t & 31;
      v8f acc = {};
      for (int k0 = 0; k0 < 512; k0 += 32) {
        v16h a = load_a_lds(c2, 512, mt * 16, k0);
        v16h bf = load_b_nk_h(w3, 512, nt * 16, k0, 512);
        acc = wmma_f16(a, bf, acc);
      }
      int n = nt * 16 + (lane & 15);
      float bi = b3[n];
#pragma unroll
      for (int r = 0; r < 8; ++r) acc[r] += bi;
      h3acc[ti] = acc;
    }
  }

  if (mode == 0) {
    int ti = 0;
    for (int t = wave; t < 64; t += 8, ++ti) {
      int nt = t & 31;
      int n = nt * 16 + (lane & 15);
      float s = 0.f, s2 = 0.f;
#pragma unroll
      for (int r = 0; r < 8; ++r) { float v = h3acc[ti][r]; s += v; s2 += v * v; }
      atomicAdd(&st512[n], s);
      atomicAdd(&st512[512 + n], s2);
    }
    return;
  }

  __syncthreads();
  {
    int ti = 0;
    for (int t = wave; t < 64; t += 8, ++ti) {
      int mt = t >> 5, nt = t & 31;
      int n = nt * 16 + (lane & 15);
      int mb = mt * 16 + ((lane >> 4) << 3);
      float mm = st512[n], iv = st512[512 + n], gg = g512[n], bb = b512[n];
#pragma unroll
      for (int r = 0; r < 8; ++r) {
        float v = (h3acc[ti][r] - mm) * iv * gg + bb;
        c2[(mb + r) * 512 + n] = (_Float16)fmaxf(v, 0.0f);
      }
    }
  }
  for (int i = tid; i < TD; i += 256) gmax4[i] = -1e30f;
  __syncthreads();

  // h4 = c2 @ w4^T + b4 (M=32,N=384,K=512), fold row-max
  for (int t = wave; t < 48; t += 8) {
    int mt = t / 24, nt = t % 24;
    v8f acc = {};
    for (int k0 = 0; k0 < 512; k0 += 32) {
      v16h a = load_a_lds(c2, 512, mt * 16, k0);
      v16h bf = load_b_nk_h(w4, 512, nt * 16, k0, TD);
      acc = wmma_f16(a, bf, acc);
    }
    int n = nt * 16 + (lane & 15);
    float bi = b4[n];
    float mx = -1e30f;
#pragma unroll
    for (int r = 0; r < 8; ++r) mx = fmaxf(mx, acc[r] + bi);
    atomicMaxF_sh(&gmax4[n], mx);
  }
  __syncthreads();
  for (int i = tid; i < TD; i += 256) tokens[grp * TD + i] = gmax4[i];
}

// ------------------------- positional embedding -----------------------------
__global__ __launch_bounds__(128) void k_pos(const float* __restrict__ centers,
                                             const float* __restrict__ w1,
                                             const float* __restrict__ b1,
                                             const float* __restrict__ w2,
                                             const float* __restrict__ b2,
                                             float* __restrict__ pos) {
  int bg = blockIdx.x;
  int b = bg / NG, g = bg % NG;
  const float* c = centers + (long)bg * 3;
  __shared__ float p1[128];
  int tid = threadIdx.x;
  const float* wr = w1 + tid * 3;
  float a = b1[tid] + wr[0] * c[0] + wr[1] * c[1] + wr[2] * c[2];
  p1[tid] = gelu_exact(a);
  __syncthreads();
  float* pr = pos + ((long)b * SEQ + 1 + g) * TD;
  for (int o = tid; o < TD; o += 128) {
    float s = b2[o];
    const float* w = w2 + (long)o * 128;
    for (int j = 0; j < 128; ++j) s += w[j] * p1[j];
    pr[o] = s;
  }
}

__global__ void k_build_x(const float* __restrict__ tokens,
                          const float* __restrict__ cls_t,
                          const float* __restrict__ cls_p,
                          float* __restrict__ x, float* __restrict__ pos) {
  long i = (long)blockIdx.x * blockDim.x + threadIdx.x;
  if (i >= (long)BB * SEQ * TD) return;
  int ch = (int)(i % TD);
  long bs = i / TD;
  int s = (int)(bs % SEQ);
  int b = (int)(bs / SEQ);
  x[i] = (s == 0) ? cls_t[ch] : tokens[((long)b * NG + s - 1) * TD + ch];
  if (s == 0) pos[i] = cls_p[ch];
}

// ------------------------------ LayerNorm (-> f16) --------------------------
__global__ __launch_bounds__(128) void k_layernorm(const float* __restrict__ x,
                                                   const float* __restrict__ g,
                                                   const float* __restrict__ b,
                                                   _Float16* __restrict__ y) {
  long row = blockIdx.x;
  const float* xr = x + row * TD;
  _Float16* yr = y + row * TD;
  __shared__ float red[128];
  int tid = threadIdx.x;
  float s = 0.f;
  for (int i = tid; i < TD; i += 128) s += xr[i];
  red[tid] = s; __syncthreads();
  for (int off = 64; off > 0; off >>= 1) { if (tid < off) red[tid] += red[tid + off]; __syncthreads(); }
  float mean = red[0] / TD; __syncthreads();
  float v = 0.f;
  for (int i = tid; i < TD; i += 128) { float d = xr[i] - mean; v += d * d; }
  red[tid] = v; __syncthreads();
  for (int off = 64; off > 0; off >>= 1) { if (tid < off) red[tid] += red[tid + off]; __syncthreads(); }
  float inv = rsqrtf(red[0] / TD + EPSF);
  for (int i = tid; i < TD; i += 128)
    yr[i] = (_Float16)((xr[i] - mean) * inv * g[i] + b[i]);
}

// ------------------------- generic WMMA GEMM (f16 in) -----------------------
// out = act(A[M,K] @ W^T + bias) (+ res). W f16 [N,K]. act: 0 none,1 relu,2 gelu.
// Writes f32 Cm and/or f16 Ch (either may be null).
__global__ __launch_bounds__(256) void k_gemm_h(const _Float16* __restrict__ A,
                                                const _Float16* __restrict__ Wt,
                                                const float* __restrict__ bias,
                                                const float* __restrict__ res,
                                                float* __restrict__ Cm,
                                                _Float16* __restrict__ Ch,
                                                int M, int K, int N, int act) {
  int tilesN = (N + 15) >> 4;
  int tilesM = (M + 15) >> 4;
  int wave = threadIdx.x >> 5;
  int tile = blockIdx.x * 8 + wave;
  if (tile >= tilesM * tilesN) return;
  int mt = tile / tilesN, nt = tile - mt * tilesN;
  int m0 = mt << 4, n0 = nt << 4;
  v8f acc = {};
  for (int k0 = 0; k0 < K; k0 += 32) {
    v16h a = load_a_h(A, K, m0, k0, M);
    v16h bf = load_b_nk_h(Wt, K, n0, k0, N);
    acc = wmma_f16(a, bf, acc);
  }
  int lane = threadIdx.x & 31;
  int n = n0 + (lane & 15);
  int mb = m0 + ((lane >> 4) << 3);
  if (n < N) {
    float bi = bias ? bias[n] : 0.0f;
#pragma unroll
    for (int r = 0; r < 8; ++r) {
      int m = mb + r;
      if (m < M) {
        float v = acc[r] + bi;
        if (act == 1) v = fmaxf(v, 0.0f);
        else if (act == 2) v = gelu_exact(v);
        if (res) v += res[(long)m * N + n];
        if (Cm) Cm[(long)m * N + n] = v;
        if (Ch) Ch[(long)m * N + n] = (_Float16)v;
      }
    }
  }
}

// ------------------------------ attention -----------------------------------
__global__ __launch_bounds__(256) void k_attn_scores(const _Float16* __restrict__ qkv,
                                                     float* __restrict__ sc) {
  constexpr int TS = (SEQ + 15) / 16;  // 17
  int wave = threadIdx.x >> 5;
  int gw = blockIdx.x * 8 + wave;
  int perBH = TS * TS;
  int bh = gw / perBH;
  if (bh >= BB * NH) return;
  int t = gw - bh * perBH;
  int mt = t / TS, nt = t - mt * TS;
  int b = bh / NH, h = bh - b * NH;
  const _Float16* qb = qkv + (long)b * SEQ * QKVD + h * HD;
  const _Float16* kb = qb + TD;
  v8f acc = {};
  for (int k0 = 0; k0 < HD; k0 += 32) {
    v16h a = load_a_h(qb, QKVD, mt * 16, k0, SEQ);
    v16h bf = load_b_nk_h(kb, QKVD, nt * 16, k0, SEQ);
    acc = wmma_f16(a, bf, acc);
  }
  float* sp = sc + (long)bh * SEQ * SEQ;
  int lane = threadIdx.x & 31;
  int n = nt * 16 + (lane & 15);
  int mb = mt * 16 + ((lane >> 4) << 3);
  if (n < SEQ) {
#pragma unroll
    for (int r = 0; r < 8; ++r) {
      int m = mb + r;
      if (m < SEQ) sp[(long)m * SEQ + n] = acc[r] * 0.125f;  // 64^-0.5
    }
  }
}

__global__ __launch_bounds__(128) void k_softmax(float* __restrict__ sc,
                                                 _Float16* __restrict__ sc16) {
  float* p = sc + (long)blockIdx.x * SEQ;
  _Float16* p16 = sc16 + (long)blockIdx.x * SEQP;
  __shared__ float red[128];
  int tid = threadIdx.x;
  float mx = -1e30f;
  for (int i = tid; i < SEQ; i += 128) mx = fmaxf(mx, p[i]);
  red[tid] = mx; __syncthreads();
  for (int off = 64; off > 0; off >>= 1) { if (tid < off) red[tid] = fmaxf(red[tid], red[tid + off]); __syncthreads(); }
  float m = red[0]; __syncthreads();
  float s = 0.f;
  for (int i = tid; i < SEQ; i += 128) { float e = __expf(p[i] - m); p[i] = e; s += e; }
  red[tid] = s; __syncthreads();
  for (int off = 64; off > 0; off >>= 1) { if (tid < off) red[tid] += red[tid + off]; __syncthreads(); }
  float inv = 1.0f / red[0];
  for (int i = tid; i < SEQ; i += 128) p16[i] = (_Float16)(p[i] * inv);
}

__global__ __launch_bounds__(256) void k_attn_out(const _Float16* __restrict__ qkv,
                                                  const _Float16* __restrict__ sc16,
                                                  _Float16* __restrict__ o) {
  constexpr int TS = (SEQ + 15) / 16;  // 17
  constexpr int TN = HD / 16;          // 4
  int wave = threadIdx.x >> 5;
  int gw = blockIdx.x * 8 + wave;
  int perBH = TS * TN;
  int bh = gw / perBH;
  if (bh >= BB * NH) return;
  int t = gw - bh * perBH;
  int mt = t / TN, nt = t - mt * TN;
  int b = bh / NH, h = bh - b * NH;
  const _Float16* sp = sc16 + (long)bh * SEQ * SEQP;
  const _Float16* vb = qkv + (long)b * SEQ * QKVD + 2 * TD + h * HD;
  v8f acc = {};
  for (int k0 = 0; k0 < SEQ; k0 += 32) {   // zero-fill past 257
    v16h a = load_a_h_bnd(sp, SEQP, mt * 16, k0, SEQ, SEQ);
    v16h bf = load_b_kn_h(vb, QKVD, nt * 16, k0, SEQ);
    acc = wmma_f16(a, bf, acc);
  }
  int lane = threadIdx.x & 31;
  int n = nt * 16 + (lane & 15);
  int mb = mt * 16 + ((lane >> 4) << 3);
#pragma unroll
  for (int r = 0; r < 8; ++r) {
    int m = mb + r;
    if (m < SEQ) o[((long)b * SEQ + m) * TD + h * HD + n] = (_Float16)acc[r];
  }
}

// ------------------------------- head ---------------------------------------
__global__ __launch_bounds__(384) void k_cls_feat(const _Float16* __restrict__ xn,
                                                  _Float16* __restrict__ cf) {
  int b = blockIdx.x, ch = threadIdx.x;
  const _Float16* xb = xn + (long)b * SEQ * TD;
  cf[(long)b * 2 * TD + ch] = xb[ch];
  float mx = -1e30f;
  for (int s = 1; s < SEQ; ++s) mx = fmaxf(mx, (float)xb[(long)s * TD + ch]);
  cf[(long)b * 2 * TD + TD + ch] = (_Float16)mx;
}

__global__ __launch_bounds__(256) void k_bn_rows(float* __restrict__ h,
                                                 const float* __restrict__ g,
                                                 const float* __restrict__ b,
                                                 int rows, int ch,
                                                 _Float16* __restrict__ out16) {
  int c = threadIdx.x;
  if (c >= ch) return;
  float s = 0.f, s2 = 0.f;
  for (int r = 0; r < rows; ++r) { float v = h[r * ch + c]; s += v; s2 += v * v; }
  float m = s / rows;
  float var = s2 / rows - m * m;
  float iv = rsqrtf(var + EPSF);
  for (int r = 0; r < rows; ++r) {
    float v = fmaxf((h[r * ch + c] - m) * iv * g[c] + b[c], 0.0f);
    h[r * ch + c] = v;
    out16[r * ch + c] = (_Float16)v;
  }
}

}  // namespace

// ------------------------------- launcher -----------------------------------
extern "C" void kernel_launch(void* const* d_in, const int* in_sizes, int n_in,
                              void* d_out, int out_size, void* d_ws, size_t ws_size,
                              hipStream_t stream) {
  (void)in_sizes; (void)n_in; (void)out_size; (void)ws_size;
  auto F = [&](int i) { return (const float*)d_in[i]; };

  char* base = (char*)d_ws;
  size_t off = 0;
  auto allocf = [&](size_t n) -> float* {
    off = (off + 255) & ~(size_t)255;
    float* p = (float*)(base + off); off += n * sizeof(float); return p;
  };
  auto alloch = [&](size_t n) -> _Float16* {
    off = (off + 255) & ~(size_t)255;
    _Float16* p = (_Float16*)(base + off); off += n * sizeof(_Float16); return p;
  };
  auto cvt = [&](const float* src, size_t n) -> const _Float16* {
    _Float16* p = alloch(n);
    k_cvt<<<(int)((n + 255) / 256), 256, 0, stream>>>(src, p, (long)n);
    return p;
  };

  // fp32 scratch
  float* fdist = allocf((size_t)BB * NP);
  int*   fidx  = (int*)allocf((size_t)BB * NG);
  float* cent  = allocf((size_t)BB * NG * 3);
  float* nb    = allocf((size_t)BB * NG * GS * CIN);
  float* h1    = allocf((size_t)BB * NG * GS * 128);
  float* st128 = allocf(256);
  float* st512 = allocf(1024);
  float* tok   = allocf((size_t)BB * NG * TD);
  float* x     = allocf((size_t)BB * SEQ * TD);
  float* pos   = allocf((size_t)BB * SEQ * TD);
  float* sc    = allocf((size_t)BB * NH * SEQ * SEQ);
  float* hh1   = allocf((size_t)BB * 256);
  float* hh2   = allocf((size_t)BB * 256);
  // f16 activations
  _Float16* xn16   = alloch((size_t)BB * SEQ * TD);
  _Float16* qkv16  = alloch((size_t)BB * SEQ * QKVD);
  _Float16* sc16   = alloch((size_t)BB * NH * SEQ * SEQP);
  _Float16* o16    = alloch((size_t)BB * SEQ * TD);
  _Float16* mlp16  = alloch((size_t)BB * SEQ * 4 * TD);
  _Float16* cf16   = alloch((size_t)BB * 2 * TD);
  _Float16* hh1h   = alloch((size_t)BB * 256);
  _Float16* hh2h   = alloch((size_t)BB * 256);

  // f16 weight pool (converted once per launch; L2-resident)
  const _Float16* w2h = cvt(F(3), (size_t)256 * 128);
  const _Float16* w3h = cvt(F(5), (size_t)512 * 512);
  const _Float16* w4h = cvt(F(7), (size_t)384 * 512);
  const _Float16 *qw[12], *pw[12], *f1w[12], *f2w[12];
  for (int blk = 0; blk < 12; ++blk) {
    int pb = 19 + blk * 11;
    qw[blk]  = cvt(F(pb + 2), (size_t)QKVD * TD);
    pw[blk]  = cvt(F(pb + 3), (size_t)TD * TD);
    f1w[blk] = cvt(F(pb + 7), (size_t)4 * TD * TD);
    f2w[blk] = cvt(F(pb + 9), (size_t)TD * 4 * TD);
  }
  const _Float16* h1w = cvt(F(153), (size_t)256 * 2 * TD);
  const _Float16* h2w = cvt(F(157), (size_t)256 * 256);
  const _Float16* h3w = cvt(F(161), (size_t)22 * 256);

  const float* pts = F(0);

  // grouping
  k_zero<<<(1280 + 255) / 256, 256, 0, stream>>>(st128, 1280);
  k_fps<<<BB, 1024, 0, stream>>>(pts, fdist, fidx);
  k_knn<<<BB * NG, 256, 0, stream>>>(pts, fidx, cent, nb);

  // encoder
  k_enc1<<<(BB * NG * GS) / 64, 128, 0, stream>>>(nb, F(1), F(2), h1, st128);
  k_bn_finalize<<<1, 128, 0, stream>>>(st128, 128, (float)(BB * NG * GS));
  k_enc_group<<<BB * NG, 256, 0, stream>>>(h1, st128, F(9), F(10), w2h, F(4),
                                           w3h, F(6), st512, nullptr, nullptr,
                                           w4h, F(8), tok, 0);
  k_bn_finalize<<<4, 128, 0, stream>>>(st512, 512, (float)(BB * NG * GS));
  k_enc_group<<<BB * NG, 256, 0, stream>>>(h1, st128, F(9), F(10), w2h, F(4),
                                           w3h, F(6), st512, F(11), F(12),
                                           w4h, F(8), tok, 1);

  // sequence + positional embedding
  k_pos<<<BB * NG, 128, 0, stream>>>(cent, F(15), F(16), F(17), F(18), pos);
  long nst = (long)BB * SEQ * TD;
  k_build_x<<<(int)((nst + 255) / 256), 256, 0, stream>>>(tok, F(13), F(14), x, pos);

  const int M = BB * SEQ;
  auto gemm = [&](const _Float16* A, const _Float16* Wt, const float* bias,
                  const float* res, float* Cm, _Float16* Ch,
                  int Mm, int K, int N, int act) {
    long tiles = (long)((Mm + 15) / 16) * ((N + 15) / 16);
    k_gemm_h<<<(int)((tiles + 7) / 8), 256, 0, stream>>>(A, Wt, bias, res, Cm, Ch,
                                                         Mm, K, N, act);
  };

  for (int blk = 0; blk < 12; ++blk) {
    int pb = 19 + blk * 11;
    k_add<<<(int)((nst + 255) / 256), 256, 0, stream>>>(x, pos, nst);
    k_layernorm<<<M, 128, 0, stream>>>(x, F(pb + 0), F(pb + 1), xn16);
    gemm(xn16, qw[blk], nullptr, nullptr, nullptr, qkv16, M, TD, QKVD, 0);
    {
      int waves = BB * NH * 17 * 17;
      k_attn_scores<<<(waves + 7) / 8, 256, 0, stream>>>(qkv16, sc);
    }
    k_softmax<<<BB * NH * SEQ, 128, 0, stream>>>(sc, sc16);
    {
      int waves = BB * NH * 17 * 4;
      k_attn_out<<<(waves + 7) / 8, 256, 0, stream>>>(qkv16, sc16, o16);
    }
    gemm(o16, pw[blk], F(pb + 4), x, x, nullptr, M, TD, TD, 0);       // x += proj(o)
    k_layernorm<<<M, 128, 0, stream>>>(x, F(pb + 5), F(pb + 6), xn16);
    gemm(xn16, f1w[blk], F(pb + 8), nullptr, nullptr, mlp16, M, TD, 4 * TD, 2);
    gemm(mlp16, f2w[blk], F(pb + 10), x, x, nullptr, M, 4 * TD, TD, 0); // x += fc2
  }

  // head
  k_layernorm<<<M, 128, 0, stream>>>(x, F(151), F(152), xn16);
  k_cls_feat<<<BB, TD, 0, stream>>>(xn16, cf16);
  gemm(cf16, h1w, F(154), nullptr, hh1, nullptr, BB, 2 * TD, 256, 0);
  k_bn_rows<<<1, 256, 0, stream>>>(hh1, F(155), F(156), BB, 256, hh1h);
  gemm(hh1h, h2w, F(158), nullptr, hh2, nullptr, BB, 256, 256, 0);
  k_bn_rows<<<1, 256, 0, stream>>>(hh2, F(159), F(160), BB, 256, hh2h);
  gemm(hh2h, h3w, F(162), nullptr, (float*)d_out, nullptr, BB, 256, 22, 0);
}
